// SelectiveWhElmanCell_2147483648058
// MI455X (gfx1250) — matrix operand order repaired and checked by reading, not compiled, not run on hardware
//
#include <hip/hip_runtime.h>
#include <hip/hip_bf16.h>
#include <math.h>

// ---------------- types for WMMA ----------------
typedef __attribute__((ext_vector_type(16))) __bf16 v16bf;
typedef __attribute__((ext_vector_type(8)))  float  v8f;

#define T_DIM 256
#define B_DIM 64
#define D_DIM 2048

#define BK 32
#define LDSK 40   // padded LDS row stride (bf16 elems): 80B rows, 16B aligned

// -------- gfx1250 async global->LDS path (guarded; fallback = sync staging) ----
#if defined(__has_builtin)
#if __has_builtin(__builtin_amdgcn_global_load_async_to_lds_b128) && \
    __has_builtin(__builtin_amdgcn_s_wait_asynccnt)
#define USE_ASYNC_LDS 1
#endif
#endif
#ifndef USE_ASYNC_LDS
#define USE_ASYNC_LDS 0
#endif

// Builtin signature (from hipcc diagnostic): (int4 AS1*, int4 AS3*, imm int, imm int)
typedef int vi4 __attribute__((ext_vector_type(4)));
typedef __attribute__((address_space(1))) vi4 gvi4;
typedef __attribute__((address_space(3))) vi4 lvi4;

__device__ __forceinline__ void async_copy16(const __bf16* g, __bf16* l) {
#if USE_ASYNC_LDS
  // generic->AS1 shares the 64-bit representation; generic LDS ptr low 32 bits == AS3 offset.
  __builtin_amdgcn_global_load_async_to_lds_b128(
      (gvi4*)(unsigned long long)(uintptr_t)g,
      (lvi4*)(unsigned int)(uintptr_t)l, 0, 0);
#else
  *reinterpret_cast<uint4*>(l) = *reinterpret_cast<const uint4*>(g);
#endif
}

__device__ __forceinline__ void wait_async_3() {
#if USE_ASYNC_LDS
  __builtin_amdgcn_s_wait_asynccnt(3);
#endif
}
__device__ __forceinline__ void wait_async_2() {
#if USE_ASYNC_LDS
  __builtin_amdgcn_s_wait_asynccnt(2);
#endif
}
__device__ __forceinline__ void wait_async_0() {
#if USE_ASYNC_LDS
  __builtin_amdgcn_s_wait_asynccnt(0);
#endif
}

// =======================================================================
// Small helper kernels (power iteration for spectral norm)
// =======================================================================
__global__ void init_u_kernel(float* __restrict__ u, int n) {
  int i = blockIdx.x * blockDim.x + threadIdx.x;
  if (i < n) {
    float s = sinf((float)i * 12.9898f + 78.233f) * 43758.5453f;
    u[i] = (s - truncf(s)) - 0.5f;   // deterministic pseudo-random init
  }
}

__global__ void normalize_kernel(float* __restrict__ v, int n) {
  __shared__ float red[256];
  float s = 0.f;
  for (int i = threadIdx.x; i < n; i += 256) { float t = v[i]; s += t * t; }
  red[threadIdx.x] = s;
  __syncthreads();
  for (int off = 128; off > 0; off >>= 1) {
    if (threadIdx.x < off) red[threadIdx.x] += red[threadIdx.x + off];
    __syncthreads();
  }
  float inv = 1.0f / (sqrtf(red[0]) + 1e-8f);
  for (int i = threadIdx.x; i < n; i += 256) v[i] *= inv;
}

__global__ void matvec_kernel(const float* __restrict__ W, const float* __restrict__ xv,
                              float* __restrict__ y, int n, int trans) {
  __shared__ float red[256];
  int row = blockIdx.x;
  float s = 0.f;
  if (trans) {
    for (int k = threadIdx.x; k < n; k += 256) s += W[(size_t)k * n + row] * xv[k];
  } else {
    for (int k = threadIdx.x; k < n; k += 256) s += W[(size_t)row * n + k] * xv[k];
  }
  red[threadIdx.x] = s;
  __syncthreads();
  for (int off = 128; off > 0; off >>= 1) {
    if (threadIdx.x < off) red[threadIdx.x] += red[threadIdx.x + off];
    __syncthreads();
  }
  if (threadIdx.x == 0) y[row] = red[0];
}

__global__ void dot_scale_kernel(const float* __restrict__ u, const float* __restrict__ w,
                                 int n, float* __restrict__ scale_out) {
  __shared__ float red[256];
  float s = 0.f;
  for (int i = threadIdx.x; i < n; i += 256) s += u[i] * w[i];
  red[threadIdx.x] = s;
  __syncthreads();
  for (int off = 128; off > 0; off >>= 1) {
    if (threadIdx.x < off) red[threadIdx.x] += red[threadIdx.x + off];
    __syncthreads();
  }
  if (threadIdx.x == 0) scale_out[0] = 0.99f / (fabsf(red[0]) + 1e-8f);
}

__global__ void convert_bf16_kernel(const float* __restrict__ src, __bf16* __restrict__ dst, size_t n) {
  for (size_t i = (size_t)blockIdx.x * blockDim.x + threadIdx.x; i < n;
       i += (size_t)gridDim.x * blockDim.x)
    dst[i] = (__bf16)src[i];
}

__global__ void scale_convert_kernel(const float* __restrict__ src, __bf16* __restrict__ dst,
                                     size_t n, const float* __restrict__ scale) {
  float s = scale[0];
  for (size_t i = (size_t)blockIdx.x * blockDim.x + threadIdx.x; i < n;
       i += (size_t)gridDim.x * blockDim.x)
    dst[i] = (__bf16)(src[i] * s);
}

__global__ void init_h_kernel(const float* __restrict__ h0, float* __restrict__ hout0,
                              __bf16* __restrict__ hbf, int n) {
  int i = blockIdx.x * blockDim.x + threadIdx.x;
  if (i < n) {
    float v = h0[i];
    hout0[i] = v;
    hbf[i] = (__bf16)v;
  }
}

// =======================================================================
// WMMA fragment loads from LDS (CDNA5 16-bit A/B layouts, wave32)
// =======================================================================
__device__ __forceinline__ v16bf load_fragA(const __bf16* row_ptr, int half) {
  v16bf a;
  uint4* q = reinterpret_cast<uint4*>(&a);
  q[0] = *reinterpret_cast<const uint4*>(row_ptr + half * 8);
  q[1] = *reinterpret_cast<const uint4*>(row_ptr + 16 + half * 8);
  return a;
}
__device__ __forceinline__ v16bf load_fragB(const __bf16* row_ptr, int half) {
  v16bf b;
  uint4* q = reinterpret_cast<uint4*>(&b);
  q[0] = *reinterpret_cast<const uint4*>(row_ptr + half * 8);
  q[1] = *reinterpret_cast<const uint4*>(row_ptr + 16 + half * 8);
  return b;
}

// =======================================================================
// Input-projection GEMM: C[m,n] = sum_k A[m,k]*W[n,k]
// Block tile 64x128, 8 waves (2 in M x 4 in N), wave tile 32x32 (4 WMMA/chunk).
// Double-buffered LDS, async global->LDS copies (ASYNCcnt pipeline).
// mode 0: outF = C + bias[n]   (xWx);   mode 1: outG = bf16(sigmoid(C))  (gate)
// =======================================================================
#define GBM 64
#define GBN 128

__global__ __launch_bounds__(256) void gemm_proj_kernel(
    const __bf16* __restrict__ A, const __bf16* __restrict__ W,
    int M, int N, int K,
    const float* __restrict__ bias,
    float* __restrict__ outF, __bf16* __restrict__ outG, int mode) {
  __shared__ __align__(16) __bf16 sA[2][GBM * LDSK];
  __shared__ __align__(16) __bf16 sB[2][GBN * LDSK];

  const int tid  = threadIdx.x;
  const int lane = tid & 31;
  const int wave = tid >> 5;
  const int waveM = wave & 1;   // m offset *32
  const int waveN = wave >> 1;  // n offset *32
  const int half = lane >> 4;
  const int l16  = lane & 15;

  const int bm = blockIdx.x * GBM;
  const int bn = blockIdx.y * GBN;

  // staging map: one b128 (8 bf16) per vector slot; 4 slots per 32-wide row
  const int srow = tid >> 2;          // 0..63
  const int scol = (tid & 3) * 8;     // 0,8,16,24

  const __bf16* gA  = A + (size_t)(bm + srow) * K + scol;
  const __bf16* gB0 = W + (size_t)(bn + srow) * K + scol;
  const __bf16* gB1 = W + (size_t)(bn + 64 + srow) * K + scol;
  const int sOff = srow * LDSK + scol;

  v8f acc00 = {0.f,0.f,0.f,0.f,0.f,0.f,0.f,0.f};
  v8f acc01 = acc00, acc10 = acc00, acc11 = acc00;

  // prologue: stage chunk 0 into buffer 0
  async_copy16(gA,        &sA[0][sOff]);
  async_copy16(gB0,       &sB[0][sOff]);
  async_copy16(gB1,       &sB[0][64 * LDSK + sOff]);

  int pb = 0;
  for (int k0 = 0; k0 < K; k0 += BK, pb ^= 1) {
    const bool more = (k0 + BK) < K;
    if (more) {
      const int nb = pb ^ 1;
      async_copy16(gA  + k0 + BK, &sA[nb][sOff]);
      async_copy16(gB0 + k0 + BK, &sB[nb][sOff]);
      async_copy16(gB1 + k0 + BK, &sB[nb][64 * LDSK + sOff]);
      wait_async_3();            // chunk k0's 3 copies complete (in-order)
    } else {
      wait_async_0();
    }
    __syncthreads();

    v16bf a0 = load_fragA(&sA[pb][(waveM * 32 +      l16) * LDSK], half);
    v16bf a1 = load_fragA(&sA[pb][(waveM * 32 + 16 + l16) * LDSK], half);
    v16bf b0 = load_fragB(&sB[pb][(waveN * 32 +      l16) * LDSK], half);
    v16bf b1 = load_fragB(&sB[pb][(waveN * 32 + 16 + l16) * LDSK], half);

    acc00 = __builtin_amdgcn_wmma_f32_16x16x32_bf16(false, a0, false, b0, (short)0, acc00, false, false);
    acc01 = __builtin_amdgcn_wmma_f32_16x16x32_bf16(false, a0, false, b1, (short)0, acc01, false, false);
    acc10 = __builtin_amdgcn_wmma_f32_16x16x32_bf16(false, a1, false, b0, (short)0, acc10, false, false);
    acc11 = __builtin_amdgcn_wmma_f32_16x16x32_bf16(false, a1, false, b1, (short)0, acc11, false, false);
    __syncthreads();             // protect buffer pb from next-next issue
  }

  // C layout: lanes 0-15 -> M=vgpr, lanes 16-31 -> M=vgpr+8; N = lane&15
  const int n0 = bn + waveN * 32 + l16;
  const int n1 = n0 + 16;
  for (int r = 0; r < 8; ++r) {
    int mloc = r + half * 8;
    size_t m0 = (size_t)(bm + waveM * 32 + mloc);
    size_t m1 = m0 + 16;
    if (mode == 0) {
      float bv0 = bias[n0], bv1 = bias[n1];
      outF[m0 * N + n0] = acc00[r] + bv0;
      outF[m0 * N + n1] = acc01[r] + bv1;
      outF[m1 * N + n0] = acc10[r] + bv0;
      outF[m1 * N + n1] = acc11[r] + bv1;
    } else {
      outG[m0 * N + n0] = (__bf16)(1.0f / (1.0f + __expf(-acc00[r])));
      outG[m0 * N + n1] = (__bf16)(1.0f / (1.0f + __expf(-acc01[r])));
      outG[m1 * N + n0] = (__bf16)(1.0f / (1.0f + __expf(-acc10[r])));
      outG[m1 * N + n1] = (__bf16)(1.0f / (1.0f + __expf(-acc11[r])));
    }
  }
}

// =======================================================================
// One recurrent step: Rh = h_prev(64xD,bf16) @ Whb^T, fused elementwise.
// Block tile 64x64 (32 blocks over N for parallelism), 8 waves 2x4,
// wave tile 32x16, async double-buffered LDS.
// =======================================================================
#define SBM 64
#define SBN 64

__global__ __launch_bounds__(256) void step_kernel(
    const __bf16* __restrict__ Hprev,   // [B][D] bf16
    const __bf16* __restrict__ Whb,     // [D][D] bf16
    const float*  __restrict__ xWx,     // [T][B][D]
    const __bf16* __restrict__ gateb,   // [T][B][D]
    const float*  __restrict__ z,       // [T][B][D]
    float* __restrict__ out,            // [T][B][D]
    float* __restrict__ hout,           // [T+1][B][D]
    __bf16* __restrict__ Hnext,         // [B][D] bf16
    int t) {
  __shared__ __align__(16) __bf16 sA[2][SBM * LDSK];
  __shared__ __align__(16) __bf16 sB[2][SBN * LDSK];

  const int tid  = threadIdx.x;
  const int lane = tid & 31;
  const int wave = tid >> 5;
  const int waveM = wave & 1;
  const int waveN = wave >> 1;
  const int half = lane >> 4;
  const int l16  = lane & 15;

  const int bn = blockIdx.x * SBN;
  const int K = D_DIM, N = D_DIM;

  const int srow = tid >> 2;
  const int scol = (tid & 3) * 8;
  const __bf16* gA = Hprev + (size_t)srow * K + scol;
  const __bf16* gB = Whb + (size_t)(bn + srow) * K + scol;
  const int sOff = srow * LDSK + scol;

  v8f acc0 = {0.f,0.f,0.f,0.f,0.f,0.f,0.f,0.f};
  v8f acc1 = acc0;

  async_copy16(gA, &sA[0][sOff]);
  async_copy16(gB, &sB[0][sOff]);

  int pb = 0;
  for (int k0 = 0; k0 < K; k0 += BK, pb ^= 1) {
    const bool more = (k0 + BK) < K;
    if (more) {
      const int nb = pb ^ 1;
      async_copy16(gA + k0 + BK, &sA[nb][sOff]);
      async_copy16(gB + k0 + BK, &sB[nb][sOff]);
      wait_async_2();
    } else {
      wait_async_0();
    }
    __syncthreads();

    v16bf a0 = load_fragA(&sA[pb][(waveM * 32 +      l16) * LDSK], half);
    v16bf a1 = load_fragA(&sA[pb][(waveM * 32 + 16 + l16) * LDSK], half);
    v16bf bb = load_fragB(&sB[pb][(waveN * 16 +      l16) * LDSK], half);

    acc0 = __builtin_amdgcn_wmma_f32_16x16x32_bf16(false, a0, false, bb, (short)0, acc0, false, false);
    acc1 = __builtin_amdgcn_wmma_f32_16x16x32_bf16(false, a1, false, bb, (short)0, acc1, false, false);
    __syncthreads();
  }

  const size_t BD = (size_t)B_DIM * D_DIM;
  const size_t tb = (size_t)t * BD;
  const int gn = bn + waveN * 16 + l16;
  for (int r = 0; r < 8; ++r) {
    int mloc = r + half * 8;
    int gm0 = waveM * 32 + mloc;
    float rh[2] = {acc0[r], acc1[r]};
    int gms[2] = {gm0, gm0 + 16};
    for (int s = 0; s < 2; ++s) {
      size_t idx = (size_t)gms[s] * N + gn;
      float xw = xWx[tb + idx];
      float g  = (float)gateb[tb + idx];
      float hn = tanhf(xw + rh[s] * g);
      float zz = z[tb + idx];
      float sz = zz / (1.0f + __expf(-zz));
      out[tb + idx]        = hn * sz;
      hout[tb + BD + idx]  = hn;        // h[t+1]
      Hnext[idx]           = (__bf16)hn;
    }
  }
}

// =======================================================================
// Host launcher
// =======================================================================
extern "C" void kernel_launch(void* const* d_in, const int* in_sizes, int n_in,
                              void* d_out, int out_size, void* d_ws, size_t ws_size,
                              hipStream_t stream) {
  (void)in_sizes; (void)n_in; (void)out_size; (void)ws_size;
  const float* x  = (const float*)d_in[0];
  const float* z  = (const float*)d_in[1];
  const float* h0 = (const float*)d_in[2];
  const float* Wx = (const float*)d_in[3];
  const float* Wh = (const float*)d_in[4];
  const float* Wg = (const float*)d_in[5];
  const float* b  = (const float*)d_in[6];

  const size_t TBD = (size_t)T_DIM * B_DIM * D_DIM;
  const size_t BD  = (size_t)B_DIM * D_DIM;
  const size_t DD  = (size_t)D_DIM * D_DIM;

  float* out  = (float*)d_out;          // [T,B,D]
  float* hout = out + TBD;              // [T+1,B,D]

  char* ws = (char*)d_ws;
  size_t off = 0;
  auto take = [&](size_t bytes) -> char* {
    char* p = ws + off;
    off = (off + bytes + 255) & ~(size_t)255;
    return p;
  };
  float*  xWx    = (float*)take(TBD * 4);
  __bf16* gateb  = (__bf16*)take(TBD * 2);
  __bf16* xb     = (__bf16*)take(TBD * 2);
  __bf16* Wxb    = (__bf16*)take(DD * 2);
  __bf16* Wgb    = (__bf16*)take(DD * 2);
  __bf16* Whb    = (__bf16*)take(DD * 2);
  __bf16* hbuf0  = (__bf16*)take(BD * 2);
  __bf16* hbuf1  = (__bf16*)take(BD * 2);
  float*  uvec   = (float*)take(D_DIM * 4);
  float*  vvec   = (float*)take(D_DIM * 4);
  float*  wvec   = (float*)take(D_DIM * 4);
  float*  scaleb = (float*)take(256);

  // ---- spectral-norm power iteration (3 steps), sigma -> scale ----
  init_u_kernel<<<(D_DIM + 255) / 256, 256, 0, stream>>>(uvec, D_DIM);
  normalize_kernel<<<1, 256, 0, stream>>>(uvec, D_DIM);
  for (int it = 0; it < 3; ++it) {
    matvec_kernel<<<D_DIM, 256, 0, stream>>>(Wh, uvec, vvec, D_DIM, 1);  // v = Wh^T u
    normalize_kernel<<<1, 256, 0, stream>>>(vvec, D_DIM);
    matvec_kernel<<<D_DIM, 256, 0, stream>>>(Wh, vvec, uvec, D_DIM, 0);  // u = Wh v
    normalize_kernel<<<1, 256, 0, stream>>>(uvec, D_DIM);
  }
  matvec_kernel<<<D_DIM, 256, 0, stream>>>(Wh, vvec, wvec, D_DIM, 0);    // w = Wh v
  dot_scale_kernel<<<1, 256, 0, stream>>>(uvec, wvec, D_DIM, scaleb);    // 0.99/(|u.w|+eps)

  // ---- bf16 conversions (8MB Wh_eff stays resident in 192MB L2 across steps) ----
  scale_convert_kernel<<<4096, 256, 0, stream>>>(Wh, Whb, DD, scaleb);
  convert_bf16_kernel<<<4096, 256, 0, stream>>>(Wx, Wxb, DD);
  convert_bf16_kernel<<<4096, 256, 0, stream>>>(Wg, Wgb, DD);
  convert_bf16_kernel<<<8192, 256, 0, stream>>>(x, xb, TBD);

  // ---- big input-projection GEMMs (WMMA bf16, f32 accumulate) ----
  dim3 gg((T_DIM * B_DIM) / GBM, D_DIM / GBN);
  gemm_proj_kernel<<<gg, 256, 0, stream>>>(xb, Wxb, T_DIM * B_DIM, D_DIM, D_DIM,
                                           b, xWx, nullptr, 0);
  gemm_proj_kernel<<<gg, 256, 0, stream>>>(xb, Wgb, T_DIM * B_DIM, D_DIM, D_DIM,
                                           nullptr, nullptr, gateb, 1);

  // ---- recurrence: h[0], then 256 sequential WMMA steps ----
  init_h_kernel<<<((int)BD + 255) / 256, 256, 0, stream>>>(h0, hout, hbuf0, (int)BD);
  for (int t = 0; t < T_DIM; ++t) {
    const __bf16* hp = (t & 1) ? hbuf1 : hbuf0;
    __bf16*       hn = (t & 1) ? hbuf0 : hbuf1;
    step_kernel<<<D_DIM / SBN, 256, 0, stream>>>(hp, Whb, xWx, gateb, z,
                                                 out, hout, hn, t);
  }
}
